// DotProductAttentionTransformerMD17_tensorserial_44212393345457
// MI455X (gfx1250) — matrix operand order repaired and checked by reading, not compiled, not run on hardware
//
#include <hip/hip_runtime.h>
#include <hip/hip_bf16.h>
#include <math.h>

// Problem constants (from reference)
#define NNODE  10000
#define NROWP  10240   // node rows padded to multiple of 256 (block M extent)
#define NEDGE  160000  // already multiple of 256
#define NGRAPH 64
#define DIM    480
#define DIMP   512     // padded feature stride (multiple of 64)
#define NHEAD  4
#define NLAYER 6
#define DHEAD  120
#define NBASIS 128
#define NSH    9
#define KCHUNK 256     // K staging chunk; LDS layout stride is fixed at KCHUNK halves
#define SMEMB  (64 * KCHUNK * 2)  // 32 KB static B-stripe buffer

typedef __attribute__((ext_vector_type(16))) __bf16 v16bf;
typedef __attribute__((ext_vector_type(8)))  __bf16 v8bf;
typedef __attribute__((ext_vector_type(8)))  float  v8f;

__device__ __forceinline__ float silu_f(float x) { return x / (1.0f + __expf(-x)); }

// ---------------------------------------------------------------------------
// Big WMMA GEMM:  C[M x Nc] = A[M x K] * B[K x Nc]
//   A  : bf16 row-major, leading dim lda
//   Bt : bf16, B transposed/col-major as [Nc][K] (leading dim K)
//   Per wave: 32(M) x 64(N) output = 2x4 fragments, 8 accumulators.
//   Block: 256 threads = 8 waves stacked in M -> block tile 256 x 64.
//   B stripe staged in LDS (fixed KCHUNK-halves column stride) via
//   global_load_async_to_lds_b128 (ASYNCcnt path); fragments read with
//   ds_load_b128 at [per-lane base + immediate offset] to avoid WMMA->VALU
//   WAR hazards on fragment registers.
//   Grid: (M/256, Nc/64). Requires M%256==0, Nc%64==0, K%32==0.
//   EPI: 0 = f32 store, 1 = bf16(silu), 2 = bf16
// ---------------------------------------------------------------------------
template<int EPI>
__global__ void k_gemm_big(const __bf16* __restrict__ A, int lda,
                           const __bf16* __restrict__ Bt, int K,
                           void* __restrict__ C, int ldc) {
  extern __shared__ __align__(16) char smem[];
  const int tid  = threadIdx.x;
  const int lane = tid & 31;
  const int wave = tid >> 5;
  const int l15  = lane & 15;
  const int half = lane >> 4;
  const int row0 = (blockIdx.x * 8 + wave) * 32;
  const int col0 = blockIdx.y * 64;

  const __bf16* __restrict__ Arow0 = A + (long)(row0 + l15) * lda + half * 8;
  const __bf16* __restrict__ Arow1 = Arow0 + (long)16 * lda;

  // per-lane LDS base for B-fragment reads; frag j / k-step kk add constant
  // offsets j*16*KCHUNK*2 + kk*2 (fits the 16-bit DS offset field)
  const char* __restrict__ bbase = smem + l15 * (KCHUNK * 2) + half * 16;

  v8f acc[2][4];
#pragma unroll
  for (int i = 0; i < 2; ++i)
#pragma unroll
    for (int j = 0; j < 4; ++j)
      acc[i][j] = (v8f){0.f, 0.f, 0.f, 0.f, 0.f, 0.f, 0.f, 0.f};

  for (int kc = 0; kc < K; kc += KCHUNK) {
    const int KC = (K - kc < KCHUNK) ? (K - kc) : KCHUNK;
    __syncthreads();                      // previous chunk fully consumed
    // ---- stage B chunk (64 cols x KC halves) into LDS, async path ----
    {
      const int segs_per_col = KC >> 3;   // 16B segments per column
      const int nseg = 64 * segs_per_col; // multiple of 256
      for (int s = tid; s < nseg; s += 256) {
        const int c  = s / segs_per_col;
        const int sg = s % segs_per_col;
        const __bf16* gsrc = Bt + (long)(col0 + c) * K + kc + sg * 8;
        unsigned ldsoff = (unsigned)(size_t)(smem + c * (KCHUNK * 2) + sg * 16);
        asm volatile("global_load_async_to_lds_b128 %0, %1, off"
                     :: "v"(ldsoff), "v"(gsrc) : "memory");
      }
      asm volatile("s_wait_asynccnt 0x0" ::: "memory");
    }
    __syncthreads();                      // all waves' async copies visible

    // ---- compute over the chunk ----
    for (int kk = 0; kk < KC; kk += 32) {
      // A fragments (global, L0/L2-cached)
      v16bf a0, a1;
      *((v8bf*)&a0)       = *(const v8bf*)(Arow0 + kc + kk);
      *(((v8bf*)&a0) + 1) = *(const v8bf*)(Arow0 + kc + kk + 16);
      *((v8bf*)&a1)       = *(const v8bf*)(Arow1 + kc + kk);
      *(((v8bf*)&a1) + 1) = *(const v8bf*)(Arow1 + kc + kk + 16);
      // all four B fragments first (distinct regs), then 8 back-to-back WMMAs
      v16bf b[4];
#pragma unroll
      for (int j = 0; j < 4; ++j) {
        const v8bf* bp = (const v8bf*)(bbase + j * (16 * KCHUNK * 2) + kk * 2);
        *((v8bf*)&b[j])       = bp[0];
        *(((v8bf*)&b[j]) + 1) = bp[2];
      }
#pragma unroll
      for (int j = 0; j < 4; ++j) {
        acc[0][j] = __builtin_amdgcn_wmma_f32_16x16x32_bf16(false, a0, false, b[j],
                                                            (short)0, acc[0][j], false, false);
        acc[1][j] = __builtin_amdgcn_wmma_f32_16x16x32_bf16(false, a1, false, b[j],
                                                            (short)0, acc[1][j], false, false);
      }
    }
  }

  // ---- epilogue ----
#pragma unroll
  for (int i = 0; i < 2; ++i) {
    const int rowBase = row0 + i * 16 + (half << 3);
#pragma unroll
    for (int j = 0; j < 4; ++j) {
      const int colOut = col0 + j * 16 + l15;
#pragma unroll
      for (int r = 0; r < 8; ++r) {
        const float v = acc[i][j][r];
        const long off = (long)(rowBase + r) * ldc + colOut;
        if (EPI == 0)      ((float*)C)[off]  = v;
        else if (EPI == 1) ((__bf16*)C)[off] = (__bf16)silu_f(v);
        else               ((__bf16*)C)[off] = (__bf16)v;
      }
    }
  }
}

// ---------------------------------------------------------------------------
// Small single-tile WMMA GEMM for narrow outputs (Nc == 16 gate heads).
// lda == K assumed. One wave per 16x16 tile.
// ---------------------------------------------------------------------------
template<int EPI>
__global__ void k_gemm16(const __bf16* __restrict__ A, const __bf16* __restrict__ Bt,
                         void* __restrict__ C, int M, int K, int Nc) {
  const int lane = threadIdx.x & 31;
  const int wave = threadIdx.x >> 5;
  const int ntn  = Nc >> 4;
  const long tile = (long)blockIdx.x * 8 + wave;
  const long total = (long)(M >> 4) * ntn;
  if (tile >= total) return;
  const int trow = (int)(tile / ntn);
  const int tcol = (int)(tile % ntn);
  const int row0 = trow << 4, col0 = tcol << 4;
  const int half = lane >> 4;
  const __bf16* Arow = A  + (long)(row0 + (lane & 15)) * K + half * 8;
  const __bf16* Brow = Bt + (long)(col0 + (lane & 15)) * K + half * 8;

  v8f acc = {0.f, 0.f, 0.f, 0.f, 0.f, 0.f, 0.f, 0.f};
  for (int k0 = 0; k0 < K; k0 += 32) {
    v16bf a, b;
    *((v8bf*)&a)       = *(const v8bf*)(Arow + k0);
    *(((v8bf*)&a) + 1) = *(const v8bf*)(Arow + k0 + 16);
    *((v8bf*)&b)       = *(const v8bf*)(Brow + k0);
    *(((v8bf*)&b) + 1) = *(const v8bf*)(Brow + k0 + 16);
    acc = __builtin_amdgcn_wmma_f32_16x16x32_bf16(false, a, false, b,
                                                  (short)0, acc, false, false);
  }
  const int rowBase = row0 + (half << 3);
  const int colOut  = col0 + (lane & 15);
#pragma unroll
  for (int r = 0; r < 8; ++r) {
    const float v = acc[r];
    const long off = (long)(rowBase + r) * Nc + colOut;
    if (EPI == 0)      ((float*)C)[off]  = v;
    else if (EPI == 1) ((__bf16*)C)[off] = (__bf16)silu_f(v);
    else               ((__bf16*)C)[off] = (__bf16)v;
  }
}

// ---------------------------------------------------------------------------
// Weight convert: src f32 [B][K][Nsrc] row-major -> dst bf16 [B][Npad][K]
// ---------------------------------------------------------------------------
__global__ void k_convw(const float* __restrict__ src, __bf16* __restrict__ dst,
                        int K, int Nsrc, int Npad, int B) {
  long idx = (long)blockIdx.x * blockDim.x + threadIdx.x;
  long tot = (long)B * Npad * K;
  if (idx >= tot) return;
  int b = (int)(idx / ((long)Npad * K));
  int r = (int)(idx % ((long)Npad * K));
  int n = r / K, k = r % K;
  float v = (n < Nsrc) ? src[(long)b * K * Nsrc + (long)k * Nsrc + n] : 0.0f;
  dst[idx] = (__bf16)v;
}

__global__ void k_conv_act(const float* __restrict__ src, __bf16* __restrict__ dst,
                           long count, float scale) {
  long i = (long)blockIdx.x * blockDim.x + threadIdx.x;
  if (i < count) dst[i] = (__bf16)(src[i] * scale);
}

__global__ void k_init_f32(float* __restrict__ p, long count, float v) {
  long i = (long)blockIdx.x * blockDim.x + threadIdx.x;
  if (i < count) p[i] = v;
}
__global__ void k_init_u32(unsigned* __restrict__ p, long count, unsigned v) {
  long i = (long)blockIdx.x * blockDim.x + threadIdx.x;
  if (i < count) p[i] = v;
}

// ---------------------------------------------------------------------------
// Per-edge geometry: spherical harmonics (E x 9, f32) + RBF (E x 128, bf16)
// ---------------------------------------------------------------------------
__global__ void k_edge_geom(const float* __restrict__ pos, const int* __restrict__ esrc,
                            const int* __restrict__ edst, float* __restrict__ sh,
                            __bf16* __restrict__ rbfb) {
  int e = blockIdx.x * blockDim.x + threadIdx.x;
  if (e >= NEDGE) return;
  int s = esrc[e], d = edst[e];
  float vx = pos[s*3+0] - pos[d*3+0];
  float vy = pos[s*3+1] - pos[d*3+1];
  float vz = pos[s*3+2] - pos[d*3+2];
  float r  = sqrtf(vx*vx + vy*vy + vz*vz);
  float rr = r + 1e-12f;
  float x = vx / rr, y = vy / rr, z = vz / rr;
  const float s3 = 1.7320508075688772f, s5 = 2.2360679774997896f, s15 = 3.8729833462074170f;
  float* o = sh + (long)e * NSH;
  o[0] = 1.0f;       o[1] = s3*x;      o[2] = s3*y;      o[3] = s3*z;
  o[4] = s15*x*y;    o[5] = s15*y*z;   o[6] = 0.5f*s5*(3.0f*z*z - 1.0f);
  o[7] = s15*x*z;    o[8] = 0.5f*s15*(x*x - y*y);
  const float width = 5.0f / 128.0f;
  const float step  = 5.0f / 127.0f;
  __bf16* rb = rbfb + (long)e * NBASIS;
#pragma unroll 8
  for (int i = 0; i < NBASIS; ++i) {
    float t = (r - step * (float)i) / width;
    rb[i] = (__bf16)__expf(-t * t);
  }
}

// x[n][d] = atom_table[node_atom[n]][d]  (row stride DIMP)
__global__ void k_embed(const float* __restrict__ atom_table, const int* __restrict__ node_atom,
                        float* __restrict__ x) {
  long idx = (long)blockIdx.x * blockDim.x + threadIdx.x;
  if (idx >= (long)NNODE * DIM) return;
  int n = (int)(idx / DIM), d = (int)(idx % DIM);
  x[(long)n * DIMP + d] = atom_table[(long)node_atom[n] * DIM + d];
}

// deg scatter: x[dst] += (sh @ Wdeg) * g0 / sqrt(AVG_DEG)
__global__ void k_deg(const float* __restrict__ sh, const float* __restrict__ Wdeg,
                      const float* __restrict__ g0, const int* __restrict__ edst,
                      float* __restrict__ x, float scale) {
  int e = blockIdx.x * blockDim.x + threadIdx.x;
  if (e >= NEDGE) return;
  const float* se = sh + (long)e * NSH;
  float s0=se[0],s1=se[1],s2=se[2],s3=se[3],s4=se[4],s5=se[5],s6=se[6],s7=se[7],s8=se[8];
  float g = g0[(long)e * 16] * scale;
  float* xr = x + (long)edst[e] * DIMP;
  for (int d = 0; d < DIM; ++d) {
    float w = s0*Wdeg[d] + s1*Wdeg[DIM+d] + s2*Wdeg[2*DIM+d] + s3*Wdeg[3*DIM+d]
            + s4*Wdeg[4*DIM+d] + s5*Wdeg[5*DIM+d] + s6*Wdeg[6*DIM+d]
            + s7*Wdeg[7*DIM+d] + s8*Wdeg[8*DIM+d];
    atomicAdd(&xr[d], w * g);
  }
}

// per-edge attention logits + segment-max via order-preserving uint key
__global__ void k_logits(const __bf16* __restrict__ XQb, const __bf16* __restrict__ XKb,
                         const float* __restrict__ sh, const float* __restrict__ Wsh_l,
                         const float* __restrict__ gate, const int* __restrict__ esrc,
                         const int* __restrict__ edst, float* __restrict__ logits,
                         unsigned* __restrict__ m, float inv) {
  int e = blockIdx.x * blockDim.x + threadIdx.x;
  if (e >= NEDGE) return;
  int s = esrc[e], dn = edst[e];
  const __bf16* q = XQb + (long)dn * DIMP;
  const __bf16* k = XKb + (long)s  * DIMP;
  const float* se = sh + (long)e * NSH;
  float s0=se[0],s1=se[1],s2=se[2],s3=se[3],s4=se[4],s5=se[5],s6=se[6],s7=se[7],s8=se[8];
#pragma unroll
  for (int h = 0; h < NHEAD; ++h) {
    float acc = 0.0f;
    for (int d0 = h * DHEAD; d0 < (h + 1) * DHEAD; d0 += 8) {
      v8bf q8 = *(const v8bf*)(q + d0);
      v8bf k8 = *(const v8bf*)(k + d0);
#pragma unroll
      for (int i = 0; i < 8; ++i) {
        int d = d0 + i;
        float kw = (float)k8[i]
                 + s0*Wsh_l[d]       + s1*Wsh_l[DIM+d]   + s2*Wsh_l[2*DIM+d]
                 + s3*Wsh_l[3*DIM+d] + s4*Wsh_l[4*DIM+d] + s5*Wsh_l[5*DIM+d]
                 + s6*Wsh_l[6*DIM+d] + s7*Wsh_l[7*DIM+d] + s8*Wsh_l[8*DIM+d];
        acc += (float)q8[i] * kw;
      }
    }
    float lg = acc * inv * gate[(long)e * 16 + h];
    logits[(long)e * NHEAD + h] = lg;
    int bi = __float_as_int(lg);
    unsigned key = (bi >= 0) ? ((unsigned)bi | 0x80000000u) : ~((unsigned)bi);
    atomicMax(&m[(long)dn * NHEAD + h], key);
  }
}

__device__ __forceinline__ float dec_key(unsigned u) {
  int bi = (u & 0x80000000u) ? (int)(u & 0x7fffffffu) : ~((int)u);
  return __int_as_float(bi);
}

// a = exp(logit - m[dst]); den[dst] += a
__global__ void k_expnorm(const float* __restrict__ logits, const unsigned* __restrict__ m,
                          const int* __restrict__ edst, float* __restrict__ aexp,
                          float* __restrict__ den) {
  long idx = (long)blockIdx.x * blockDim.x + threadIdx.x;
  if (idx >= (long)NEDGE * NHEAD) return;
  int e = (int)(idx >> 2), h = (int)(idx & 3);
  int dn = edst[e];
  float mm = dec_key(m[(long)dn * NHEAD + h]);
  float a = __expf(logits[idx] - mm);
  aexp[idx] = a;
  atomicAdd(&den[(long)dn * NHEAD + h], a);
}

// agg[dst] += alpha * v[src]
__global__ void k_msg(const float* __restrict__ aexp, const float* __restrict__ den,
                      const __bf16* __restrict__ XVb, const int* __restrict__ esrc,
                      const int* __restrict__ edst, float* __restrict__ agg) {
  int e = blockIdx.x * blockDim.x + threadIdx.x;
  if (e >= NEDGE) return;
  int s = esrc[e], dn = edst[e];
  float al[NHEAD];
#pragma unroll
  for (int h = 0; h < NHEAD; ++h)
    al[h] = aexp[(long)e * NHEAD + h] / (den[(long)dn * NHEAD + h] + 1e-12f);
  const __bf16* v = XVb + (long)s * DIMP;
  float* ar = agg + (long)dn * DIMP;
#pragma unroll
  for (int h = 0; h < NHEAD; ++h) {
    float a = al[h];
    for (int d0 = h * DHEAD; d0 < (h + 1) * DHEAD; d0 += 8) {
      v8bf v8 = *(const v8bf*)(v + d0);
#pragma unroll
      for (int i = 0; i < 8; ++i) atomicAdd(&ar[d0 + i], a * (float)v8[i]);
    }
  }
}

// x = LayerNorm(x + T); refresh bf16 shadow. One wave32 per node (480 = 32*15).
__global__ void k_ln_res(float* __restrict__ x, const float* __restrict__ T,
                         __bf16* __restrict__ xb) {
  int lane = threadIdx.x & 31;
  int node = blockIdx.x * 8 + (threadIdx.x >> 5);
  if (node >= NNODE) return;
  const int d0 = lane * 15;
  float* xr = x + (long)node * DIMP;
  const float* tr = T + (long)node * DIMP;
  float vals[15];
  float sum = 0.f, ss = 0.f;
#pragma unroll
  for (int i = 0; i < 15; ++i) {
    float v = xr[d0 + i] + tr[d0 + i];
    vals[i] = v; sum += v; ss += v * v;
  }
#pragma unroll
  for (int o = 16; o > 0; o >>= 1) {
    sum += __shfl_xor(sum, o);
    ss  += __shfl_xor(ss, o);
  }
  float mu = sum * (1.0f / DIM);
  float var = ss * (1.0f / DIM) - mu * mu;
  float rs = rsqrtf(var + 1e-5f);
  __bf16* xbr = xb + (long)node * DIMP;
#pragma unroll
  for (int i = 0; i < 15; ++i) {
    float o = (vals[i] - mu) * rs;
    xr[d0 + i] = o;
    xbr[d0 + i] = (__bf16)o;
  }
}

// energy[batch[n]] += dot(silu_h[n], Wh2) / AVG_NODES   (T3b row stride DIMP)
__global__ void k_energy(const __bf16* __restrict__ T3b, const float* __restrict__ Wh2,
                         const int* __restrict__ batch, float* __restrict__ out, float invn) {
  int n = blockIdx.x * blockDim.x + threadIdx.x;
  if (n >= NNODE) return;
  const __bf16* t = T3b + (long)n * DIMP;
  float acc = 0.f;
  for (int d0 = 0; d0 < DIM; d0 += 8) {
    v8bf t8 = *(const v8bf*)(t + d0);
#pragma unroll
    for (int i = 0; i < 8; ++i) acc += (float)t8[i] * Wh2[d0 + i];
  }
  atomicAdd(&out[batch[n]], acc * invn);
}

// ---------------------------------------------------------------------------
extern "C" void kernel_launch(void* const* d_in, const int* in_sizes, int n_in,
                              void* d_out, int out_size, void* d_ws, size_t ws_size,
                              hipStream_t stream) {
  const float* pos        = (const float*)d_in[0];
  const int*   node_atom  = (const int*)d_in[1];
  const int*   esrc       = (const int*)d_in[2];
  const int*   edst       = (const int*)d_in[3];
  const int*   batch      = (const int*)d_in[4];
  const float* atom_table = (const float*)d_in[5];
  const float* Wdeg       = (const float*)d_in[6];
  const float* Wd1        = (const float*)d_in[7];
  const float* Wd2        = (const float*)d_in[8];
  const float* Wd3        = (const float*)d_in[9];
  const float* Wq         = (const float*)d_in[10];
  const float* Wk         = (const float*)d_in[11];
  const float* Wv         = (const float*)d_in[12];
  const float* Wsh        = (const float*)d_in[13];
  const float* W1         = (const float*)d_in[14];
  const float* W2         = (const float*)d_in[15];
  const float* W3         = (const float*)d_in[16];
  const float* Wo         = (const float*)d_in[17];
  const float* Wf1        = (const float*)d_in[18];
  const float* Wf2        = (const float*)d_in[19];
  const float* Wh1        = (const float*)d_in[20];
  const float* Wh2        = (const float*)d_in[21];
  float* out = (float*)d_out;

  char* wsb = (char*)d_ws;
  size_t off = 0;
  auto alloc = [&](size_t bytes) -> void* {
    void* p = wsb + off;
    off += (bytes + 255) & ~(size_t)255;
    return p;
  };

  // bf16 transposed (and N-padded) weights
  __bf16* wd1T = (__bf16*)alloc((size_t)64 * 128 * 2);
  __bf16* wd2T = (__bf16*)alloc((size_t)64 * 64 * 2);
  __bf16* wd3T = (__bf16*)alloc((size_t)16 * 64 * 2);
  __bf16* wqT  = (__bf16*)alloc((size_t)NLAYER * DIMP * DIM * 2);
  __bf16* wkT  = (__bf16*)alloc((size_t)NLAYER * DIMP * DIM * 2);
  __bf16* wvT  = (__bf16*)alloc((size_t)NLAYER * DIMP * DIM * 2);
  __bf16* woT  = (__bf16*)alloc((size_t)NLAYER * DIMP * DIM * 2);
  __bf16* w1T  = (__bf16*)alloc((size_t)NLAYER * 64 * 128 * 2);
  __bf16* w2T  = (__bf16*)alloc((size_t)NLAYER * 64 * 64 * 2);
  __bf16* w3T  = (__bf16*)alloc((size_t)NLAYER * 16 * 64 * 2);
  __bf16* wf1T = (__bf16*)alloc((size_t)NLAYER * 960 * DIM * 2);
  __bf16* wf2T = (__bf16*)alloc((size_t)NLAYER * DIMP * 960 * 2);
  __bf16* wh1T = (__bf16*)alloc((size_t)DIMP * DIM * 2);
  // activations / edge buffers
  float*  shbuf = (float*)alloc((size_t)NEDGE * NSH * 4);
  __bf16* rbfb  = (__bf16*)alloc((size_t)NEDGE * NBASIS * 2);
  float*  X     = (float*)alloc((size_t)NROWP * DIMP * 4);
  __bf16* Xb    = (__bf16*)alloc((size_t)NROWP * DIMP * 2);
  __bf16* XQb   = (__bf16*)alloc((size_t)NROWP * DIMP * 2);
  __bf16* XKb   = (__bf16*)alloc((size_t)NROWP * DIMP * 2);
  __bf16* XVb   = (__bf16*)alloc((size_t)NROWP * DIMP * 2);
  __bf16* H1b   = (__bf16*)alloc((size_t)NEDGE * 64 * 2);
  __bf16* H2b   = (__bf16*)alloc((size_t)NEDGE * 64 * 2);
  float*  Gbuf  = (float*)alloc((size_t)NEDGE * 16 * 4);
  float*  logits= (float*)alloc((size_t)NEDGE * NHEAD * 4);
  float*  aexp  = (float*)alloc((size_t)NEDGE * NHEAD * 4);
  unsigned* mbuf= (unsigned*)alloc((size_t)NNODE * NHEAD * 4);
  float*  den   = (float*)alloc((size_t)NNODE * NHEAD * 4);
  float*  agg   = (float*)alloc((size_t)NROWP * DIMP * 4);
  __bf16* AGGb  = (__bf16*)alloc((size_t)NROWP * DIMP * 2);
  float*  Tbuf  = (float*)alloc((size_t)NROWP * DIMP * 4);
  __bf16* T1b   = (__bf16*)alloc((size_t)NROWP * 960 * 2);
  (void)ws_size; (void)n_in; (void)in_sizes; (void)out_size;

  const float inv_sqrt_deg = (float)(1.0 / sqrt(15.57930850982666));
  const float inv_dh       = (float)(1.0 / sqrt((double)DHEAD));
  const float inv_nodes    = (float)(1.0 / 18.03065905448718);

  auto convw = [&](const float* src, __bf16* dst, int K, int Nsrc, int Npad, int B) {
    long tot = (long)B * Npad * K;
    k_convw<<<(int)((tot + 255) / 256), 256, 0, stream>>>(src, dst, K, Nsrc, Npad, B);
  };
  // big GEMM: M%256==0, Nc%64==0, K%32==0
  auto gemmB = [&](const __bf16* A, int lda, const __bf16* Bt, int K,
                   void* C, int ldc, int M, int Nc, int epi) {
    dim3 grid(M / 256, Nc / 64);
    if (epi == 0)      k_gemm_big<0><<<grid, 256, SMEMB, stream>>>(A, lda, Bt, K, C, ldc);
    else if (epi == 1) k_gemm_big<1><<<grid, 256, SMEMB, stream>>>(A, lda, Bt, K, C, ldc);
    else               k_gemm_big<2><<<grid, 256, SMEMB, stream>>>(A, lda, Bt, K, C, ldc);
  };
  auto gemmS = [&](const __bf16* A, const __bf16* Bt, void* C, int M, int K, int Nc) {
    long tiles = (long)(M / 16) * (Nc / 16);
    k_gemm16<0><<<(int)((tiles + 7) / 8), 256, 0, stream>>>(A, Bt, C, M, K, Nc);
  };
  auto initf = [&](float* p, long n, float v) {
    k_init_f32<<<(int)((n + 255) / 256), 256, 0, stream>>>(p, n, v);
  };

  // ---- weight conversion ----
  convw(Wd1, wd1T, 128, 64, 64, 1);
  convw(Wd2, wd2T, 64, 64, 64, 1);
  convw(Wd3, wd3T, 64, 1, 16, 1);
  convw(Wq,  wqT,  DIM, DIM, DIMP, NLAYER);
  convw(Wk,  wkT,  DIM, DIM, DIMP, NLAYER);
  convw(Wv,  wvT,  DIM, DIM, DIMP, NLAYER);
  convw(Wo,  woT,  DIM, DIM, DIMP, NLAYER);
  convw(W1,  w1T,  128, 64, 64, NLAYER);
  convw(W2,  w2T,  64, 64, 64, NLAYER);
  convw(W3,  w3T,  64, NHEAD, 16, NLAYER);
  convw(Wf1, wf1T, DIM, 960, 960, NLAYER);
  convw(Wf2, wf2T, 960, DIM, DIMP, NLAYER);
  convw(Wh1, wh1T, DIM, DIM, DIMP, 1);

  const int EB  = (NEDGE + 255) / 256;
  const int NDB = (int)(((long)NROWP * DIMP + 255) / 256);
  const int NEB = (int)(((long)NNODE * DIM + 255) / 256);

  // ---- edge geometry + rbf ----
  k_edge_geom<<<EB, 256, 0, stream>>>(pos, esrc, edst, shbuf, rbfb);

  // ---- degree-embedding gate MLP (WMMA) ----
  gemmB(rbfb, 128, wd1T, 128, H1b, 64, NEDGE, 64, 1);
  gemmB(H1b,  64,  wd2T, 64,  H2b, 64, NEDGE, 64, 1);
  gemmS(H2b, wd3T, Gbuf, NEDGE, 64, 16);

  // ---- node init ----
  k_embed<<<NEB, 256, 0, stream>>>(atom_table, node_atom, X);
  k_deg<<<EB, 256, 0, stream>>>(shbuf, Wdeg, Gbuf, edst, X, inv_sqrt_deg);
  k_conv_act<<<NDB, 256, 0, stream>>>(X, Xb, (long)NROWP * DIMP, 1.0f);

  // ---- transformer layers ----
  for (int l = 0; l < NLAYER; ++l) {
    const __bf16* wq_l  = wqT  + (size_t)l * DIMP * DIM;
    const __bf16* wk_l  = wkT  + (size_t)l * DIMP * DIM;
    const __bf16* wv_l  = wvT  + (size_t)l * DIMP * DIM;
    const __bf16* wo_l  = woT  + (size_t)l * DIMP * DIM;
    const __bf16* w1_l  = w1T  + (size_t)l * 64 * 128;
    const __bf16* w2_l  = w2T  + (size_t)l * 64 * 64;
    const __bf16* w3_l  = w3T  + (size_t)l * 16 * 64;
    const __bf16* wf1_l = wf1T + (size_t)l * 960 * DIM;
    const __bf16* wf2_l = wf2T + (size_t)l * DIMP * 960;
    const float*  wsh_l = Wsh + (size_t)l * NSH * DIM;

    // projections -> bf16 (L2-resident for edge gathers)
    gemmB(Xb, DIMP, wq_l, DIM, XQb, DIMP, NROWP, DIMP, 2);
    gemmB(Xb, DIMP, wk_l, DIM, XKb, DIMP, NROWP, DIMP, 2);
    gemmB(Xb, DIMP, wv_l, DIM, XVb, DIMP, NROWP, DIMP, 2);

    // per-edge gate MLP (WMMA)
    gemmB(rbfb, 128, w1_l, 128, H1b, 64, NEDGE, 64, 1);
    gemmB(H1b,  64,  w2_l, 64,  H2b, 64, NEDGE, 64, 1);
    gemmS(H2b, w3_l, Gbuf, NEDGE, 64, 16);

    // segment softmax
    k_init_u32<<<(int)(((long)NNODE * NHEAD + 255) / 256), 256, 0, stream>>>(
        mbuf, (long)NNODE * NHEAD, 0u);
    k_logits<<<EB, 256, 0, stream>>>(XQb, XKb, shbuf, wsh_l, Gbuf, esrc, edst,
                                     logits, mbuf, inv_dh);
    initf(den, (long)NNODE * NHEAD, 0.0f);
    k_expnorm<<<(int)(((long)NEDGE * NHEAD + 255) / 256), 256, 0, stream>>>(
        logits, mbuf, edst, aexp, den);
    initf(agg, (long)NROWP * DIMP, 0.0f);
    k_msg<<<EB, 256, 0, stream>>>(aexp, den, XVb, esrc, edst, agg);

    // attention output projection + residual LN
    k_conv_act<<<NDB, 256, 0, stream>>>(agg, AGGb, (long)NROWP * DIMP, inv_sqrt_deg);
    gemmB(AGGb, DIMP, wo_l, DIM, Tbuf, DIMP, NROWP, DIMP, 0);
    k_ln_res<<<(NNODE + 7) / 8, 256, 0, stream>>>(X, Tbuf, Xb);

    // feed-forward + residual LN
    gemmB(Xb, DIMP, wf1_l, DIM, T1b, 960, NROWP, 960, 1);
    gemmB(T1b, 960, wf2_l, 960, Tbuf, DIMP, NROWP, DIMP, 0);
    k_ln_res<<<(NNODE + 7) / 8, 256, 0, stream>>>(X, Tbuf, Xb);
  }

  // ---- readout ----
  gemmB(Xb, DIMP, wh1T, DIM, T1b, DIMP, NROWP, DIMP, 1);
  initf(out, NGRAPH, 0.0f);
  k_energy<<<(NNODE + 255) / 256, 256, 0, stream>>>(T1b, Wh2, batch, out, inv_nodes);
}
